// Quantization_80728205295803
// MI455X (gfx1250) — matrix operand order repaired and checked by reading, not compiled, not run on hardware
//
#include <hip/hip_runtime.h>
#include <hip/hip_fp16.h>

// ---------------------------------------------------------------------------
// VQ codebook quantization for MI455X (gfx1250, wave32, WMMA).
//   dist(i,k) = ||x_i||^2 + ||c_k||^2 - 2 x_i . c_k
// argmin over k is invariant to the per-row ||x_i||^2 term, so the scored
// quantity is  cnorm[k] - 2 * (X @ C^T)[i,k]  with the GEMM done with
// v_wmma_f32_16x16x32_f16 and cnorm computed in f32 from the f16-rounded
// codebook for numerical consistency. loss / emb_out are recomputed exactly
// in f32 from the winning code.
//
// vq_argmin layout (per 256-thread WG, 8 waves):
//   - 128 x-rows staged f32->f16 in LDS (64 KB), A fragments hoisted to
//     registers (8 x v16h per wave), then the same 64 KB LDS is reused as a
//     double-buffered 2x32KB codebook chunk (64 codes each).
//   - wave w exclusively owns rows [16w,16w+16): 4 column tiles x 8 k-steps
//     = 32 WMMAs per chunk between barriers; argmin reduction is wave-local.
//   - next chunk staged while computing: CDNA5 async global->LDS copy
//     (ASYNCcnt) when available, else register prefetch + ds_store.
// ---------------------------------------------------------------------------

typedef __attribute__((ext_vector_type(16))) _Float16 v16h;
typedef __attribute__((ext_vector_type(8)))  _Float16 v8h;
typedef __attribute__((ext_vector_type(4)))  _Float16 v4h;
typedef __attribute__((ext_vector_type(8)))  float    v8f;
typedef __attribute__((ext_vector_type(4)))  int      v4i;

// async copy operand types: AS(1) = global ("__device__"), AS(3) = LDS
typedef __attribute__((address_space(1))) v4i gv4i;
typedef __attribute__((address_space(3))) v4i lv4i;

#define NROWS 65536
#define DIM   256
#define KCODE 1024
#define MT    128            // x rows per workgroup
#define NC    64             // codebook codes per LDS chunk (32 KB)
#define NCHUNK (KCODE / NC)  // 16
#define CHUNK_BYTES (NC * DIM * 2)   // 32768
#define LDSBYTES 65536

#if defined(__AMDGCN__) && __has_builtin(__builtin_amdgcn_global_load_async_to_lds_b128)
#define ASYNC_LDS 1
#else
#define ASYNC_LDS 0
#endif

__device__ __forceinline__ void wait_async0() {
#if __has_builtin(__builtin_amdgcn_s_wait_asynccnt)
  __builtin_amdgcn_s_wait_asynccnt(0);
#else
  asm volatile("s_wait_asynccnt 0" ::: "memory");
#endif
}

// ---------------------------------------------------------------------------
// Kernel A: codebook f32 -> f16 (workspace) + cnorm[k] = sum(f16(c)^2) in f32.
// ---------------------------------------------------------------------------
__global__ __launch_bounds__(64) void vq_prep(const float* __restrict__ cb,
                                              _Float16* __restrict__ cb16,
                                              float* __restrict__ cnorm) {
  const int k = blockIdx.x;
  const int t = threadIdx.x;
  const float4* src = (const float4*)(cb + (size_t)k * DIM);
  float4 v = src[t];
  v4h hv;
  hv[0] = (_Float16)v.x; hv[1] = (_Float16)v.y;
  hv[2] = (_Float16)v.z; hv[3] = (_Float16)v.w;
  *(v4h*)(cb16 + (size_t)k * DIM + t * 4) = hv;
  float s = (float)hv[0]*(float)hv[0] + (float)hv[1]*(float)hv[1]
          + (float)hv[2]*(float)hv[2] + (float)hv[3]*(float)hv[3];
  #pragma unroll
  for (int m = 16; m >= 1; m >>= 1) s += __shfl_xor(s, m, 32);
  __shared__ float part[2];
  if ((t & 31) == 0) part[t >> 5] = s;
  __syncthreads();
  if (t == 0) cnorm[k] = part[0] + part[1];
}

// ---------------------------------------------------------------------------
// Kernel B: WMMA argmin. NROWS/128 WGs x 256 threads (8 waves).
// ---------------------------------------------------------------------------
__global__ __launch_bounds__(256) void vq_argmin(const float* __restrict__ x,
                                                 const _Float16* __restrict__ cb16,
                                                 const float* __restrict__ cnorm,
                                                 int* __restrict__ ids) {
  extern __shared__ char smem[];                 // 64 KB dynamic
  const int tid = threadIdx.x;
  const int bid = blockIdx.x;
  const int rowBase = bid * MT;

  // ---- phase 1: stage 128x256 x tile f32 -> f16 into LDS ----
  {
    _Float16* xlds = (_Float16*)smem;            // [128][256]
    const float4* xs = (const float4*)(x + (size_t)rowBase * DIM);
    #pragma unroll
    for (int i = 0; i < 32; ++i) {
      int flat = tid + i * 256;                  // 8192 float4 in tile
      int r  = flat >> 6;                        // 64 float4 per row
      int c4 = flat & 63;
      float4 v = xs[(size_t)r * 64 + c4];
      v4h hv;
      hv[0] = (_Float16)v.x; hv[1] = (_Float16)v.y;
      hv[2] = (_Float16)v.z; hv[3] = (_Float16)v.w;
      *(v4h*)&xlds[r * DIM + c4 * 4] = hv;
    }
  }
  __syncthreads();

  const int wave = tid >> 5, lane = tid & 31;
  const int g = lane >> 4, lm = lane & 15;
  const int mR = wave * 16;                      // wave-exclusive 16 rows

  // ---- hoist A fragments: CDNA5 16-bit A 16x32 layout: lane M = l&15;
  // halves 0..7 = K[8g..8g+8), halves 8..15 = K[16+8g..16+8g+8) ----
  v16h afrag[8];
  {
    const _Float16* xlds = (const _Float16*)smem;
    #pragma unroll
    for (int ks = 0; ks < 8; ++ks) {
      const int kk = ks * 32;
      union { v16h v; v8h h[2]; } u;
      u.h[0] = *(const v8h*)&xlds[(mR + lm) * DIM + kk + 8 * g];
      u.h[1] = *(const v8h*)&xlds[(mR + lm) * DIM + kk + 16 + 8 * g];
      afrag[ks] = u.v;
    }
  }
  __syncthreads();   // all waves done reading x tile; LDS now reused for codes

  // ---- phase 2: LDS = double-buffered codebook chunk [2][64][256] f16 ----
  _Float16* clds0 = (_Float16*)smem;
  _Float16* clds1 = (_Float16*)(smem + CHUNK_BYTES);

  float minv[8];
  int   mini[8];
  #pragma unroll
  for (int r = 0; r < 8; ++r) { minv[r] = 3.4e38f; mini[r] = 0x7fffffff; }

  // chunk is contiguous in cb16 -> staging is a flat linear 32 KB copy:
  // 2048 x 16B, 8 per thread.
#if ASYNC_LDS
  {
    char* g0 = (char*)cb16;
    char* l0 = (char*)clds0;
    #pragma unroll
    for (int i = 0; i < 8; ++i) {
      int off = (tid + i * 256) * 16;
      __builtin_amdgcn_global_load_async_to_lds_b128(
          (gv4i*)(g0 + off), (lv4i*)(l0 + off), 0, 0);
    }
    wait_async0();
  }
  __syncthreads();
#else
  {
    const uint4* gs = (const uint4*)cb16;
    uint4* ls = (uint4*)clds0;
    #pragma unroll
    for (int i = 0; i < 8; ++i) {
      int u = tid + i * 256;
      ls[u] = gs[u];
    }
  }
  __syncthreads();
#endif

  int buf = 0;
  for (int c = 0; c < NCHUNK; ++c) {
    const int nb = c * NC;
    _Float16* cur = buf ? clds1 : clds0;
    _Float16* nxt = buf ? clds0 : clds1;

    // ---- issue next-chunk staging (overlaps with WMMA below) ----
#if ASYNC_LDS
    if (c + 1 < NCHUNK) {
      char* gn = (char*)(cb16 + (size_t)(nb + NC) * DIM);
      char* ln = (char*)nxt;
      #pragma unroll
      for (int i = 0; i < 8; ++i) {
        int off = (tid + i * 256) * 16;
        __builtin_amdgcn_global_load_async_to_lds_b128(
            (gv4i*)(gn + off), (lv4i*)(ln + off), 0, 0);
      }
    }
#else
    uint4 pf[8];
    if (c + 1 < NCHUNK) {
      const uint4* gn = (const uint4*)(cb16 + (size_t)(nb + NC) * DIM);
      #pragma unroll
      for (int i = 0; i < 8; ++i) pf[i] = gn[tid + i * 256];
    }
#endif

    // ---- 4 column tiles x 8 k-steps = 32 WMMAs per wave per chunk ----
    // CDNA5 16-bit B 32x16 layout: lane N = l&15; halves 0..7 = K[8g..8g+8),
    // halves 8..15 = K[16+8g..16+8g+8) -> two 16B LDS reads per fragment.
    #pragma unroll
    for (int t = 0; t < 4; ++t) {
      const int code = t * 16 + lm;
      const _Float16* crow = cur + code * DIM;
      v8f acc = {};
      #pragma unroll
      for (int ks = 0; ks < 8; ++ks) {
        const int kk = ks * 32;
        union { v16h v; v8h h[2]; } b;
        b.h[0] = *(const v8h*)(crow + kk + 8 * g);
        b.h[1] = *(const v8h*)(crow + kk + 16 + 8 * g);
        acc = __builtin_amdgcn_wmma_f32_16x16x32_f16(
            false, afrag[ks], false, b.v, (short)0, acc, false, false);
      }
      // C/D layout: VGPR r, lane l -> row r+8g, col l&15 of the tile.
      // Tiles scanned in ascending code order; strict '<' keeps lowest id.
      const int gcode = nb + code;
      const float cn = cnorm[gcode];
      #pragma unroll
      for (int r = 0; r < 8; ++r) {
        float d = cn - 2.0f * acc[r];
        if (d < minv[r]) { minv[r] = d; mini[r] = gcode; }
      }
    }

    // ---- make next buffer ready, single barrier per chunk ----
#if ASYNC_LDS
    wait_async0();
#else
    if (c + 1 < NCHUNK) {
      uint4* ln = (uint4*)nxt;
      #pragma unroll
      for (int i = 0; i < 8; ++i) ln[tid + i * 256] = pf[i];
    }
#endif
    __syncthreads();
    buf ^= 1;
  }

  // ---- wave-local reduce across the 16 lanes (columns) per row ----
  #pragma unroll
  for (int r = 0; r < 8; ++r) {
    float v = minv[r]; int ix = mini[r];
    #pragma unroll
    for (int m = 8; m >= 1; m >>= 1) {
      float ov = __shfl_xor(v, m, 32);
      int   oi = __shfl_xor(ix, m, 32);
      if (ov < v || (ov == v && oi < ix)) { v = ov; ix = oi; }
    }
    minv[r] = v; mini[r] = ix;
  }
  if (lm == 0) {                     // lane 0 -> rows mR+0..7, lane 16 -> +8..15
    const int rb = rowBase + mR + 8 * g;
    #pragma unroll
    for (int r = 0; r < 8; ++r) ids[rb + r] = mini[r];
  }
}

// ---------------------------------------------------------------------------
// Kernel C: gather winner, exact f32 loss = 1.25*||x-c||^2, emb_out = x+(c-x),
// ids as float. One wave per row, 8 floats (2 float4) per lane.
// ---------------------------------------------------------------------------
__global__ __launch_bounds__(32) void vq_out(const float* __restrict__ x,
                                             const float* __restrict__ cb,
                                             const int* __restrict__ ids,
                                             float* __restrict__ emb_out,
                                             float* __restrict__ idf,
                                             float* __restrict__ loss) {
  const int row = blockIdx.x, t = threadIdx.x;
  const int id = ids[row];
  const float4* xr = (const float4*)(x  + (size_t)row * DIM);
  const float4* cr = (const float4*)(cb + (size_t)id  * DIM);
  float4* er = (float4*)(emb_out + (size_t)row * DIM);
  float s = 0.f;
  #pragma unroll
  for (int i = 0; i < 2; ++i) {
    int c = t + i * 32;
    float4 xv = xr[c], cv = cr[c];
    float4 ev;
    ev.x = xv.x + (cv.x - xv.x);  // match reference x + (emb - x) rounding
    ev.y = xv.y + (cv.y - xv.y);
    ev.z = xv.z + (cv.z - xv.z);
    ev.w = xv.w + (cv.w - xv.w);
    er[c] = ev;
    float dx = xv.x - cv.x, dy = xv.y - cv.y, dz = xv.z - cv.z, dw = xv.w - cv.w;
    s += dx*dx + dy*dy + dz*dz + dw*dw;
  }
  #pragma unroll
  for (int m = 16; m >= 1; m >>= 1) s += __shfl_xor(s, m, 32);
  if (t == 0) { loss[row] = 1.25f * s; idf[row] = (float)id; }
}

// ---------------------------------------------------------------------------
extern "C" void kernel_launch(void* const* d_in, const int* in_sizes, int n_in,
                              void* d_out, int out_size, void* d_ws, size_t ws_size,
                              hipStream_t stream) {
  const float* x  = (const float*)d_in[0];   // [65536, 256] f32
  const float* cb = (const float*)d_in[1];   // [1024, 256] f32
  float* out = (float*)d_out;                // emb_out | ids(as f32) | loss

  // workspace layout
  _Float16* cb16  = (_Float16*)d_ws;                                   // 512 KB
  float*    cnorm = (float*)((char*)d_ws + (size_t)KCODE * DIM * 2);   // 4 KB
  int*      ids   = (int*)((char*)cnorm + (size_t)KCODE * 4);          // 256 KB

  float* emb_out = out;
  float* idf     = out + (size_t)NROWS * DIM;
  float* loss    = out + (size_t)NROWS * DIM + NROWS;

  vq_prep  <<<KCODE,      64, 0,        stream>>>(cb, cb16, cnorm);
  vq_argmin<<<NROWS / MT, 256, LDSBYTES, stream>>>(x, cb16, cnorm, ids);
  vq_out   <<<NROWS,      32, 0,        stream>>>(x, cb, ids, emb_out, idf, loss);
}
